// KroneckerFusion_14577119003160
// MI455X (gfx1250) — matrix operand order repaired and checked by reading, not compiled, not run on hardware
//
#include <hip/hip_runtime.h>
#include <hip/hip_bf16.h>

typedef __attribute__((ext_vector_type(16))) __bf16 v16bf;
typedef __attribute__((ext_vector_type(8)))  float  v8f;
typedef __attribute__((ext_vector_type(4)))  unsigned int v4u;
typedef __attribute__((ext_vector_type(8)))  int v8i;
typedef __attribute__((ext_vector_type(4)))  int v4i;
typedef __attribute__((address_space(1))) v4i g_v4i;   // global int4
typedef __attribute__((address_space(3))) v4i l_v4i;   // LDS int4
typedef __attribute__((address_space(3))) char as3char;

#ifndef __has_builtin
#define __has_builtin(x) 0
#endif
#if __has_builtin(__builtin_amdgcn_global_load_async_to_lds_b128)
#define HAVE_ASYNC 1
#else
#define HAVE_ASYNC 0
#endif
#if __has_builtin(__builtin_amdgcn_tensor_load_to_lds) && __has_builtin(__builtin_amdgcn_s_wait_tensorcnt)
#define HAVE_TDM 1
#else
#define HAVE_TDM 0
#endif

// ---------------- workspace layout (halves) ----------------
#define OFF_SMALL 0        // 8 tiles * 512 halves (Bh1 x2, Bh2 x2, Bo1 x2, Bo2 x2)
#define OFF_BZ    4096     // 32 ksteps * 4 ntiles * 512
#define OFF_BE1   69632    // 34 ksteps * 4 ntiles * 512
#define OFF_BE2   139264   // 4  ksteps * 4 ntiles * 512
#define WS_HALVES 147456
#define BIAS_FLOATS 320

// ---------------- LDS layout (bytes) ----------------
#define LDS_SMALL 0                     // 8192 B
#define LDS_WE2   8192                  // 16384 B
#define LDS_BIG   24576                 // 139264 B (Bz then BWe1)
#define LDS_WAVE  (24576 + 139264)      // per-wave stash
#define WBUF_F    576                   // 16x36 floats
#define NWAVES    8
#define NBUF      6                     // tmp0 tmp1 o1s[0] o2s[0] o1s[1] o2s[1]
#define TILES     2
#define LDS_TOTAL (LDS_WAVE + NWAVES*NBUF*WBUF_F*4)   // 274432 B (< 320KB WGP LDS)

__device__ inline v8f vzero() {
  v8f z;
#pragma unroll
  for (int i = 0; i < 8; ++i) z[i] = 0.f;
  return z;
}

__device__ inline v8f wmma_bf16(v16bf a, v16bf b, v8f c) {
  return __builtin_amdgcn_wmma_f32_16x16x32_bf16(false, a, false, b, (short)0, c,
                                                 false, false);
}

__device__ inline v16bf bfrag(const __bf16* base, int tile, int lane) {
  return *(const v16bf*)(base + tile * 512 + lane * 16);
}

__device__ inline v16bf patFrag(const float* buf, int m, int kb) {
  const float* row = buf + m * 36;
  float t[16];
  *(float4*)(t + 0)  = *(const float4*)(row + kb);
  *(float4*)(t + 4)  = *(const float4*)(row + kb + 4);
  *(float4*)(t + 8)  = *(const float4*)(row + kb + 16);
  *(float4*)(t + 12) = *(const float4*)(row + kb + 20);
  v16bf a;
#pragma unroll
  for (int e = 0; e < 16; ++e) a[e] = (__bf16)t[e];
  return a;
}

__device__ inline void loadPatF32(float* dst, const float* row, int kb) {
  *(float4*)(dst + 0)  = *(const float4*)(row + kb);
  *(float4*)(dst + 4)  = *(const float4*)(row + kb + 4);
  *(float4*)(dst + 8)  = *(const float4*)(row + kb + 16);
  *(float4*)(dst + 12) = *(const float4*)(row + kb + 20);
}

__device__ inline void storeD(float* buf, v8f c, int nt, int lane) {
  int n = nt * 16 + (lane & 15);
  int hi = lane >> 4;
#pragma unroll
  for (int r = 0; r < 8; ++r) buf[(r + 8 * hi) * 36 + n] = c[r];
}

// ---- gfx1250 data movers ----
__device__ inline void wait_async() {
#if HAVE_ASYNC
#if __has_builtin(__builtin_amdgcn_s_wait_asynccnt)
  __builtin_amdgcn_s_wait_asynccnt(0);
#else
  asm volatile("s_wait_asynccnt 0x0" ::: "memory");
#endif
#endif
}

__device__ inline void stage_async(const __bf16* src, __bf16* dst, int n16, int tid) {
#if HAVE_ASYNC
  g_v4i* s = (g_v4i*)src;
  l_v4i* d = (l_v4i*)dst;
  for (int i = tid; i < n16; i += 256)
    __builtin_amdgcn_global_load_async_to_lds_b128(s + i, d + i, 0, 0);
#else
  const uint4* s = (const uint4*)src;
  uint4* d = (uint4*)dst;
  for (int i = tid; i < n16; i += 256) d[i] = s[i];
#endif
}

// One-shot 1-D TDM copy global->LDS (issued by one wave; tracked by TENSORcnt)
__device__ inline void stage_big(const __bf16* src, __bf16* dst, unsigned bytes,
                                 int tid, int wave) {
#if HAVE_TDM
  if (wave == 0) {
    unsigned long long ga = (unsigned long long)(size_t)src;
    unsigned la = (unsigned)(size_t)(as3char*)(dst);
    unsigned n8 = bytes >> 3;                       // 8-byte elements
    v4u g0;
    g0.x = 1u;                                      // count=1 (valid descriptor)
    g0.y = la;                                      // lds_addr
    g0.z = (unsigned)ga;                            // global_addr[31:0]
    g0.w = (unsigned)((ga >> 32) & 0x01FFFFFFu) | (2u << 30);  // addr hi + type=2
    v8i g1;
    g1[0] = (int)(3u << 16);                        // data_size = 8B
    g1[1] = (int)((n8 & 0xFFFFu) << 16);            // tensor_dim0 [15:0]
    g1[2] = (int)(((n8 >> 16) & 0xFFFFu) | (1u << 16)); // tensor_dim0 hi | tensor_dim1=1
    g1[3] = (int)((n8 & 0xFFFFu) << 16);            // tile_dim0 = n8
    g1[4] = 0;                                      // tile_dim1/2 unused
    g1[5] = (int)n8;                                // tensor_dim0_stride lo
    g1[6] = 0; g1[7] = 0;
    v4i gz = {0, 0, 0, 0};
#if __clang_major__ >= 23
    v8i gz8 = {0, 0, 0, 0, 0, 0, 0, 0};
    __builtin_amdgcn_tensor_load_to_lds(g0, g1, gz, gz, gz8, 0);
#else
    __builtin_amdgcn_tensor_load_to_lds(g0, g1, gz, gz, 0);
#endif
    __builtin_amdgcn_s_wait_tensorcnt(0);
  }
#else
  const uint4* s = (const uint4*)src;
  uint4* d = (uint4*)dst;
  int n16 = (int)(bytes >> 4);
  for (int i = tid; i < n16; i += 256) d[i] = s[i];
#endif
}

// =====================  weight prep: fp32 -> packed bf16 B-fragment images =====================
__global__ void kf_prep(const float* __restrict__ Wh1, const float* __restrict__ Wh2,
                        const float* __restrict__ Wo1, const float* __restrict__ Wo2,
                        const float* __restrict__ Wz1, const float* __restrict__ Wz2,
                        const float* __restrict__ We1, const float* __restrict__ We2,
                        const float* __restrict__ bh1, const float* __restrict__ bh2,
                        const float* __restrict__ bz1, const float* __restrict__ bz2,
                        const float* __restrict__ bo1, const float* __restrict__ bo2,
                        const float* __restrict__ be1, const float* __restrict__ be2,
                        __bf16* __restrict__ wsH, float* __restrict__ wsB) {
  int idx = blockIdx.x * blockDim.x + threadIdx.x;
  if (idx < WS_HALVES) {
    int lane = (idx >> 4) & 31;
    int e = idx & 15;
    int k = (e & 7) + ((lane >> 4) << 3) + ((e >> 3) << 4);   // B-frag K
    int nl = lane & 15;
    float val;
    if (idx < OFF_BZ) {
      int t = idx >> 9;
      int mat = t >> 1, nt = t & 1;
      const float* W = (mat == 0) ? Wh1 : (mat == 1) ? Wh2 : (mat == 2) ? Wo1 : Wo2;
      int n = nt * 16 + nl;
      val = W[n * 32 + k];
    } else if (idx < OFF_BE1) {
      int g = (idx - OFF_BZ) >> 9;
      int kc = g >> 2, nt = g & 3;
      int n = nt * 16 + nl;
      val = (n < 32) ? Wz1[n * 1024 + kc * 32 + k]
                     : Wz2[(n - 32) * 1024 + kc * 32 + k];
    } else if (idx < OFF_BE2) {
      int g = (idx - OFF_BE1) >> 9;
      int c = g >> 2, nt = g & 3;
      int n = nt * 16 + nl;
      int col = (c < 33) ? (c * 33 + k) : (k * 33 + 32);
      val = We1[n * 1089 + col];
    } else {
      int g = (idx - OFF_BE2) >> 9;
      int ks = g >> 2, nt = g & 3;
      int n = nt * 16 + nl;
      int kg = ks * 32 + k;
      int col = (kg < 96) ? kg : kg + 1;
      val = We2[n * 130 + col];
    }
    wsH[idx] = (__bf16)val;
  } else {
    int b = idx - WS_HALVES;
    if (b < 32)        wsB[b] = bh1[b];
    else if (b < 64)   wsB[b] = bh2[b - 32];
    else if (b < 96)   wsB[b] = bz1[b - 64];
    else if (b < 128)  wsB[b] = bz2[b - 96];
    else if (b < 160)  wsB[b] = bo1[b - 128];
    else if (b < 192)  wsB[b] = bo2[b - 160];
    else if (b < 256) { int n = b - 192; wsB[b] = be1[n] + We1[n * 1089 + 1088]; }
    else if (b < BIAS_FLOATS) {
      int n = b - 256; wsB[b] = be2[n] + We2[n * 130 + 96] + We2[n * 130 + 129];
    }
  }
}

// =====================  fused main kernel  =====================
__global__ void __launch_bounds__(256) kf_fused(const float* __restrict__ x,
                                                const __bf16* __restrict__ wsH,
                                                const float* __restrict__ wsB,
                                                float* __restrict__ out, int Btot) {
  extern __shared__ char smem[];
  __bf16* sSmall = (__bf16*)(smem + LDS_SMALL);
  __bf16* sWe2   = (__bf16*)(smem + LDS_WE2);
  __bf16* sBig   = (__bf16*)(smem + LDS_BIG);
  float*  sWave  = (float*)(smem + LDS_WAVE);

  const int tid  = threadIdx.x;
  const int lane = tid & 31;
  const int wave = tid >> 5;
  const int m    = lane & 15;
  const int hiA  = lane >> 4;
  const int kb   = hiA * 8;

  // ---- stage small Bs + BWe2 (async-to-LDS) and Bz (TDM) ----
  stage_async(wsH + OFF_SMALL, sSmall, 512, tid);
  stage_async(wsH + OFF_BE2, sWe2, 1024, tid);
  stage_big(wsH + OFF_BZ, sBig, 131072u, tid, wave);
  wait_async();
  __syncthreads();

  float* wbase = sWave + wave * NBUF * WBUF_F;
  float* tmp0 = wbase;
  float* tmp1 = wbase + WBUF_F;

  const int blockRow0 = blockIdx.x * (NWAVES * 16 * TILES);

  // =========== front half (needs Bz): h, z, gate, o for both tiles ===========
#pragma unroll 1
  for (int t = 0; t < TILES; ++t) {
    float* o1s = wbase + (2 + 2 * t) * WBUF_F;
    float* o2s = o1s + WBUF_F;
    const int row0 = blockRow0 + wave * (16 * TILES) + t * 16;

    const float* v1p = x + row0 * 32 + m * 32;
    const float* v2p = x + Btot * 32 + row0 * 32 + m * 32;
    alignas(16) float rv1[16], rv2[16];
    loadPatF32(rv1, v1p, kb);
    loadPatF32(rv2, v2p, kb);

    // stash v1 rows for per-chunk scalars
    *(float4*)(tmp0 + m * 36 + kb)      = *(const float4*)(rv1 + 0);
    *(float4*)(tmp0 + m * 36 + kb + 4)  = *(const float4*)(rv1 + 4);
    *(float4*)(tmp0 + m * 36 + kb + 16) = *(const float4*)(rv1 + 8);
    *(float4*)(tmp0 + m * 36 + kb + 20) = *(const float4*)(rv1 + 12);

    v16bf aV1, aV2;
#pragma unroll
    for (int e = 0; e < 16; ++e) { aV1[e] = (__bf16)rv1[e]; aV2[e] = (__bf16)rv2[e]; }

    v8f h1a[2], h2a[2];
#pragma unroll
    for (int nt = 0; nt < 2; ++nt) {
      h1a[nt] = wmma_bf16(aV1, bfrag(sSmall, 0 + nt, lane), vzero());
      h2a[nt] = wmma_bf16(aV2, bfrag(sSmall, 2 + nt, lane), vzero());
    }

    // bilinear z1||z2 = kron(v1,v2) @ Bz
    v8f za[4];
#pragma unroll
    for (int nt = 0; nt < 4; ++nt) za[nt] = vzero();
    for (int kc = 0; kc < 32; ++kc) {
      float s = tmp0[m * 36 + kc];
      v16bf aK;
#pragma unroll
      for (int e = 0; e < 16; ++e) aK[e] = (__bf16)(rv2[e] * s);
#pragma unroll
      for (int nt = 0; nt < 4; ++nt)
        za[nt] = wmma_bf16(aK, bfrag(sBig, kc * 4 + nt, lane), za[nt]);
    }

    // activations + gate; stash g1->tmp0, g2->tmp1
#pragma unroll
    for (int nt = 0; nt < 2; ++nt) {
      float b1 = wsB[0 + nt * 16 + m], b2 = wsB[32 + nt * 16 + m];
#pragma unroll
      for (int r = 0; r < 8; ++r) {
        h1a[nt][r] = fmaxf(h1a[nt][r] + b1, 0.f);
        h2a[nt][r] = fmaxf(h2a[nt][r] + b2, 0.f);
      }
    }
#pragma unroll
    for (int nt = 0; nt < 2; ++nt) {
      float bzv1 = wsB[64 + nt * 16 + m], bzv2 = wsB[96 + nt * 16 + m];
      int n = nt * 16 + m;
#pragma unroll
      for (int r = 0; r < 8; ++r) {
        float z1 = za[nt][r] + bzv1;
        float z2 = za[2 + nt][r] + bzv2;
        float g1 = h1a[nt][r] / (1.f + __expf(-z1));
        float g2 = h2a[nt][r] / (1.f + __expf(-z2));
        tmp0[(r + 8 * hiA) * 36 + n] = g1;
        tmp1[(r + 8 * hiA) * 36 + n] = g2;
      }
    }

    // o1/o2 GEMMs; stash rows + ones column
    v16bf aG1 = patFrag(tmp0, m, kb);
    v16bf aG2 = patFrag(tmp1, m, kb);
#pragma unroll
    for (int nt = 0; nt < 2; ++nt) {
      v8f o1 = wmma_bf16(aG1, bfrag(sSmall, 4 + nt, lane), vzero());
      v8f o2 = wmma_bf16(aG2, bfrag(sSmall, 6 + nt, lane), vzero());
      float b1 = wsB[128 + nt * 16 + m], b2 = wsB[160 + nt * 16 + m];
#pragma unroll
      for (int r = 0; r < 8; ++r) {
        o1[r] = fmaxf(o1[r] + b1, 0.f);
        o2[r] = fmaxf(o2[r] + b2, 0.f);
      }
      storeD(o1s, o1, nt, lane);
      storeD(o2s, o2, nt, lane);
    }
    if (hiA == 0) { o1s[m * 36 + 32] = 1.f; o2s[m * 36 + 32] = 1.f; }
  }

  // ---- swap LDS big region: Bz -> BWe1 (TDM) ----
  __syncthreads();
  stage_big(wsH + OFF_BE1, sBig, 139264u, tid, wave);
  __syncthreads();

  // =========== back half (needs BWe1): e1, e2 for both tiles ===========
#pragma unroll 1
  for (int t = 0; t < TILES; ++t) {
    float* o1s = wbase + (2 + 2 * t) * WBUF_F;
    float* o2s = o1s + WBUF_F;
    const int row0 = blockRow0 + wave * (16 * TILES) + t * 16;

    alignas(16) float ro2[16];
    loadPatF32(ro2, o2s + m * 36, kb);

    // out = relu(kron(o1,o2) @ We1^T + be1')
    v8f ea[4];
#pragma unroll
    for (int nt = 0; nt < 4; ++nt) ea[nt] = vzero();
    for (int c = 0; c < 33; ++c) {
      float s = o1s[m * 36 + c];          // includes o1[:,32]==1
      v16bf aK;
#pragma unroll
      for (int e = 0; e < 16; ++e) aK[e] = (__bf16)(ro2[e] * s);
#pragma unroll
      for (int nt = 0; nt < 4; ++nt)
        ea[nt] = wmma_bf16(aK, bfrag(sBig, c * 4 + nt, lane), ea[nt]);
    }
    {
      v16bf aO1 = patFrag(o1s, m, kb);    // chunk 33: o1[0..31]
#pragma unroll
      for (int nt = 0; nt < 4; ++nt)
        ea[nt] = wmma_bf16(aO1, bfrag(sBig, 33 * 4 + nt, lane), ea[nt]);
    }
#pragma unroll
    for (int nt = 0; nt < 4; ++nt) {
      float bb = wsB[192 + nt * 16 + m];
      v8f v = ea[nt];
#pragma unroll
      for (int r = 0; r < 8; ++r) v[r] = fmaxf(v[r] + bb, 0.f);
      storeD((nt < 2) ? tmp0 : tmp1, v, nt & 1, lane);   // stash 'out' rows
    }

    // out2 = relu([out|o1|o2] @ We2^T + be2')
    v8f fa[4];
#pragma unroll
    for (int nt = 0; nt < 4; ++nt) fa[nt] = vzero();
    const float* ksrc[4] = {tmp0, tmp1, o1s, o2s};
#pragma unroll
    for (int ks = 0; ks < 4; ++ks) {
      v16bf aF = patFrag(ksrc[ks], m, kb);
#pragma unroll
      for (int nt = 0; nt < 4; ++nt)
        fa[nt] = wmma_bf16(aF, bfrag(sWe2, ks * 4 + nt, lane), fa[nt]);
    }
#pragma unroll
    for (int nt = 0; nt < 4; ++nt) {
      float bb = wsB[256 + nt * 16 + m];
      int n = nt * 16 + m;
#pragma unroll
      for (int r = 0; r < 8; ++r) {
        float v = fmaxf(fa[nt][r] + bb, 0.f);
        out[(row0 + r + 8 * hiA) * 64 + n] = v;
      }
    }
  }
}

extern "C" void kernel_launch(void* const* d_in, const int* in_sizes, int n_in,
                              void* d_out, int out_size, void* d_ws, size_t ws_size,
                              hipStream_t stream) {
  const float* x   = (const float*)d_in[0];
  const float* Wh1 = (const float*)d_in[1];
  const float* bh1 = (const float*)d_in[2];
  const float* Wz1 = (const float*)d_in[3];
  const float* bz1 = (const float*)d_in[4];
  const float* Wo1 = (const float*)d_in[5];
  const float* bo1 = (const float*)d_in[6];
  const float* Wh2 = (const float*)d_in[7];
  const float* bh2 = (const float*)d_in[8];
  const float* Wz2 = (const float*)d_in[9];
  const float* bz2 = (const float*)d_in[10];
  const float* Wo2 = (const float*)d_in[11];
  const float* bo2 = (const float*)d_in[12];
  const float* We1 = (const float*)d_in[13];
  const float* be1 = (const float*)d_in[14];
  const float* We2 = (const float*)d_in[15];
  const float* be2 = (const float*)d_in[16];
  float* out = (float*)d_out;

  __bf16* wsH = (__bf16*)d_ws;
  float*  wsB = (float*)((char*)d_ws + (size_t)WS_HALVES * 2);

  int Btot = in_sizes[0] / 64;   // x is [2, B, 32]

  (void)hipFuncSetAttribute((const void*)kf_fused,
                            hipFuncAttributeMaxDynamicSharedMemorySize, LDS_TOTAL);

  int prepN = WS_HALVES + BIAS_FLOATS;
  kf_prep<<<(prepN + 255) / 256, 256, 0, stream>>>(
      Wh1, Wh2, Wo1, Wo2, Wz1, Wz2, We1, We2,
      bh1, bh2, bz1, bz2, bo1, bo2, be1, be2, wsH, wsB);

  kf_fused<<<Btot / (NWAVES * 16 * TILES), 256, LDS_TOTAL, stream>>>(x, wsH, wsB, out, Btot);
}
